// GQA_72172630442441
// MI455X (gfx1250) — matrix operand order repaired and checked by reading, not compiled, not run on hardware
//
#include <hip/hip_runtime.h>

typedef _Float16 v16h __attribute__((ext_vector_type(16)));
typedef _Float16 v8h  __attribute__((ext_vector_type(8)));
typedef _Float16 v4h  __attribute__((ext_vector_type(4)));
typedef float    v8f  __attribute__((ext_vector_type(8)));
typedef float    v4f  __attribute__((ext_vector_type(4)));
typedef unsigned int v8u __attribute__((ext_vector_type(8)));
typedef unsigned int v2u __attribute__((ext_vector_type(2)));

#define NB   2
#define NQH  32
#define NKH  8
#define NS   2048
#define ND   128
#define GRP  (NQH / NKH)

#define QTILE 128       // queries per workgroup (8 waves x 16)
#define KTILE 64        // keys per inner iteration (4 x 16-key tiles)
#define NTHREADS 256

#define K_STRIDE 136    // halfs per K-tile row (128 + 8 pad); 272B, 16B-aligned
#define V_STRIDE 72     // halfs per Vt row (64 + 8 pad); 144B, 16B-aligned

#define SHUF16(lo, hi) __builtin_shufflevector((lo), (hi), 0,1,2,3,4,5,6,7,8,9,10,11,12,13,14,15)

// v_cvt_pk_rtz_f16_f32: convert+pack two f32 -> one u32 of two f16
static __device__ __forceinline__ unsigned cvt_pk_u32(float a, float b) {
    return __builtin_bit_cast(unsigned, __builtin_amdgcn_cvt_pkrtz(a, b));
}
static __device__ __forceinline__ v4h pack4(float a, float b, float c, float d) {
    v2u w; w[0] = cvt_pk_u32(a, b); w[1] = cvt_pk_u32(c, d);
    return __builtin_bit_cast(v4h, w);
}

__global__ __launch_bounds__(NTHREADS)
void gqa_flash_wmma(const float* __restrict__ Q,
                    const float* __restrict__ K,
                    const float* __restrict__ V,
                    float* __restrict__ O)
{
    // double-buffered staging tiles
    __shared__ _Float16 Klds[2][KTILE * K_STRIDE];  // key-major f16 (A-operand for S^T)
    __shared__ _Float16 Vt[2][ND * V_STRIDE];       // d-major (transposed) f16 (A-operand for O^T)

    const int tid  = threadIdx.x;
    const int wave = tid >> 5;
    const int lane = tid & 31;
    const int ln   = lane & 15;       // query column (B/C lane dim) or matrix row (A)
    const int hi   = lane >> 4;       // which 16-lane half

    const int qtile = blockIdx.x;     // S / QTILE
    const int qh    = blockIdx.y;
    const int b     = blockIdx.z;
    const int kh    = qh / GRP;

    const int q0 = qtile * QTILE + wave * 16;

    // log2(e) / sqrt(128): scores land directly in the exp2 domain
    const float softmax_scale = 0.12751744752f;

    const float* Qg = Q + (((size_t)b * NQH + qh) * NS + q0) * ND;
    const float* Kg = K + (((size_t)b * NKH + kh) * NS) * ND;
    const float* Vg = V + (((size_t)b * NKH + kh) * NS) * ND;
    float*       Og = O + (((size_t)b * NQH + qh) * NS + q0) * ND;

    // per-thread staging indices (constant across iterations)
    const int k_key[8] = { (tid + 0*NTHREADS) >> 5, (tid + 1*NTHREADS) >> 5,
                           (tid + 2*NTHREADS) >> 5, (tid + 3*NTHREADS) >> 5,
                           (tid + 4*NTHREADS) >> 5, (tid + 5*NTHREADS) >> 5,
                           (tid + 6*NTHREADS) >> 5, (tid + 7*NTHREADS) >> 5 };
    const int dg0 = tid & 31;                      // same low bits for every i
    const int v_kg[2] = { tid >> 5, (tid + NTHREADS) >> 5 };

    // ---- Q as WMMA B-operand (Q^T, 32x16 per d-chunk), scale folded in ----
    v16h qB[4];
    {
        const float* qrow = Qg + (size_t)ln * ND;
        #pragma unroll
        for (int c = 0; c < 4; ++c) {
            const int base = c * 32 + hi * 16;
            #pragma unroll
            for (int e = 0; e < 16; ++e)
                qB[c][e] = (_Float16)(qrow[base + e] * softmax_scale);
        }
    }

    const v8f zero8 = {0.f, 0.f, 0.f, 0.f, 0.f, 0.f, 0.f, 0.f};
    v8f acc[8];                       // O^T: chunk n holds d rows n*16..n*16+15, lane=query
    #pragma unroll
    for (int n = 0; n < 8; ++n) acc[n] = zero8;

    float m = -1e30f, l = 0.0f;       // per-lane (= per-query) softmax state (log2 domain)

    const int nblocks = NS / KTILE;

    float4 fk[8], fv[8];              // in-flight staging registers (software pipeline)

    // ---- prologue: load + stage tile 0 ----
    {
        const float4* src  = (const float4*)Kg;
        const float4* vsrc = (const float4*)Vg;
        #pragma unroll
        for (int i = 0; i < 8; ++i) fk[i] = src[tid + i * NTHREADS];
        #pragma unroll
        for (int i = 0; i < 2; ++i)
            #pragma unroll
            for (int j = 0; j < 4; ++j) fv[i * 4 + j] = vsrc[(v_kg[i] * 4 + j) * 32 + dg0];

        #pragma unroll
        for (int i = 0; i < 8; ++i)
            *(v4h*)&Klds[0][k_key[i] * K_STRIDE + dg0 * 4] = pack4(fk[i].x, fk[i].y, fk[i].z, fk[i].w);
        #pragma unroll
        for (int i = 0; i < 2; ++i) {
            float4 f0 = fv[i*4+0], f1 = fv[i*4+1], f2 = fv[i*4+2], f3 = fv[i*4+3];
            *(v4h*)&Vt[0][(dg0 * 4 + 0) * V_STRIDE + v_kg[i] * 4] = pack4(f0.x, f1.x, f2.x, f3.x);
            *(v4h*)&Vt[0][(dg0 * 4 + 1) * V_STRIDE + v_kg[i] * 4] = pack4(f0.y, f1.y, f2.y, f3.y);
            *(v4h*)&Vt[0][(dg0 * 4 + 2) * V_STRIDE + v_kg[i] * 4] = pack4(f0.z, f1.z, f2.z, f3.z);
            *(v4h*)&Vt[0][(dg0 * 4 + 3) * V_STRIDE + v_kg[i] * 4] = pack4(f0.w, f1.w, f2.w, f3.w);
        }
    }
    __syncthreads();

    for (int kb = 0; kb < nblocks; ++kb) {
        const int cur = kb & 1;
        const int nxt = cur ^ 1;
        const bool have_next = (kb + 1 < nblocks);

        // ---- issue next tile's global loads early (latency hides under compute) ----
        if (have_next) {
            const float4* src  = (const float4*)(Kg + (size_t)(kb + 1) * KTILE * ND);
            const float4* vsrc = (const float4*)(Vg + (size_t)(kb + 1) * KTILE * ND);
            #pragma unroll
            for (int i = 0; i < 8; ++i) fk[i] = src[tid + i * NTHREADS];
            #pragma unroll
            for (int i = 0; i < 2; ++i)
                #pragma unroll
                for (int j = 0; j < 4; ++j) fv[i * 4 + j] = vsrc[(v_kg[i] * 4 + j) * 32 + dg0];
            if (kb + 2 < nblocks) {
                __builtin_prefetch(Kg + (size_t)(kb + 2) * KTILE * ND + tid * 32, 0, 3);
                __builtin_prefetch(Vg + (size_t)(kb + 2) * KTILE * ND + tid * 32, 0, 3);
            }
        }

        // ---- S^T tiles: st[t] = K[t*16..t*16+15] @ Q^T  (keys in regs, queries in lanes)
        v8f st[4];
        #pragma unroll
        for (int t = 0; t < 4; ++t) st[t] = zero8;
        #pragma unroll
        for (int c = 0; c < 4; ++c) {
            #pragma unroll
            for (int t = 0; t < 4; ++t) {
                const _Float16* kp = &Klds[cur][(t * 16 + ln) * K_STRIDE + c * 32];
                v16h aK = SHUF16(*(const v8h*)(kp + hi * 16),
                                 *(const v8h*)(kp + 8 + hi * 16));
                st[t] = __builtin_amdgcn_wmma_f32_16x16x32_f16(false, aK, false, qB[c],
                                                               (short)0, st[t], false, false);
            }
        }

        // ---- online softmax, per-lane (query) over 64 keys, exp2 domain ----
        float vmax = -1e30f;
        #pragma unroll
        for (int t = 0; t < 4; ++t)
            #pragma unroll
            for (int r = 0; r < 8; ++r)
                vmax = fmaxf(vmax, st[t][r]);
        vmax = fmaxf(vmax, __shfl_xor(vmax, 16, 32));   // combine key halves

        const float mnew = fmaxf(m, vmax);
        if (__any(vmax > m)) {                           // rescale only when max moved
            const float alpha = __builtin_amdgcn_exp2f(m - mnew);
            #pragma unroll
            for (int n = 0; n < 8; ++n)
                #pragma unroll
                for (int r = 0; r < 8; ++r) acc[n][r] *= alpha;
            l *= alpha;
        }
        m = mnew;

        float ls = 0.0f;
        #pragma unroll
        for (int t = 0; t < 4; ++t)
            #pragma unroll
            for (int r = 0; r < 8; ++r) {
                const float p = __builtin_amdgcn_exp2f(st[t][r] - m);
                st[t][r] = p;
                ls += p;
            }
        ls += __shfl_xor(ls, 16, 32);
        l += ls;

        // ---- pack P^T to f16 and build B-operands (keys contiguous per lane) ----
        unsigned pk0[4], pk1[4], pk2[4], pk3[4];
        #pragma unroll
        for (int j = 0; j < 4; ++j) {
            pk0[j] = cvt_pk_u32(st[0][2*j], st[0][2*j+1]);
            pk1[j] = cvt_pk_u32(st[1][2*j], st[1][2*j+1]);
            pk2[j] = cvt_pk_u32(st[2][2*j], st[2][2*j+1]);
            pk3[j] = cvt_pk_u32(st[3][2*j], st[3][2*j+1]);
        }
        v8u b0w, b1w;
        #pragma unroll
        for (int j = 0; j < 4; ++j) {
            const unsigned prov0 = hi ? pk0[j] : pk1[j];
            const unsigned got0  = (unsigned)__shfl_xor((int)prov0, 16, 32);
            b0w[j]     = hi ? got0 : pk0[j];
            b0w[j + 4] = hi ? pk1[j] : got0;
            const unsigned prov1 = hi ? pk2[j] : pk3[j];
            const unsigned got1  = (unsigned)__shfl_xor((int)prov1, 16, 32);
            b1w[j]     = hi ? got1 : pk2[j];
            b1w[j + 4] = hi ? pk3[j] : got1;
        }
        const v16h B0 = __builtin_bit_cast(v16h, b0w);
        const v16h B1 = __builtin_bit_cast(v16h, b1w);

        // ---- O^T += V^T @ P^T : 8 d-chunks, K=64 as two K=32 WMMAs ----
        #pragma unroll
        for (int n = 0; n < 8; ++n) {
            const _Float16* vp = &Vt[cur][(n * 16 + ln) * V_STRIDE];
            v16h a0 = SHUF16(*(const v8h*)(vp + hi * 16),
                             *(const v8h*)(vp + 8 + hi * 16));
            acc[n] = __builtin_amdgcn_wmma_f32_16x16x32_f16(false, a0, false, B0,
                                                            (short)0, acc[n], false, false);
            v16h a1 = SHUF16(*(const v8h*)(vp + 32 + hi * 16),
                             *(const v8h*)(vp + 40 + hi * 16));
            acc[n] = __builtin_amdgcn_wmma_f32_16x16x32_f16(false, a1, false, B1,
                                                            (short)0, acc[n], false, false);
        }

        // ---- convert + store next tile into the other buffer, single barrier ----
        if (have_next) {
            #pragma unroll
            for (int i = 0; i < 8; ++i)
                *(v4h*)&Klds[nxt][k_key[i] * K_STRIDE + dg0 * 4] =
                    pack4(fk[i].x, fk[i].y, fk[i].z, fk[i].w);
            #pragma unroll
            for (int i = 0; i < 2; ++i) {
                float4 f0 = fv[i*4+0], f1 = fv[i*4+1], f2 = fv[i*4+2], f3 = fv[i*4+3];
                *(v4h*)&Vt[nxt][(dg0 * 4 + 0) * V_STRIDE + v_kg[i] * 4] = pack4(f0.x, f1.x, f2.x, f3.x);
                *(v4h*)&Vt[nxt][(dg0 * 4 + 1) * V_STRIDE + v_kg[i] * 4] = pack4(f0.y, f1.y, f2.y, f3.y);
                *(v4h*)&Vt[nxt][(dg0 * 4 + 2) * V_STRIDE + v_kg[i] * 4] = pack4(f0.z, f1.z, f2.z, f3.z);
                *(v4h*)&Vt[nxt][(dg0 * 4 + 3) * V_STRIDE + v_kg[i] * 4] = pack4(f0.w, f1.w, f2.w, f3.w);
            }
            __syncthreads();
        }
    }

    // ---- epilogue: O^T reg r = d (n*16 + r + 8*hi), lane = query ln ----
    const float invl = 1.0f / l;
    #pragma unroll
    for (int n = 0; n < 8; ++n) {
        v4f lo4, hi4;
        #pragma unroll
        for (int r = 0; r < 4; ++r) { lo4[r] = acc[n][r] * invl; hi4[r] = acc[n][r + 4] * invl; }
        float* op = &Og[(size_t)ln * ND + n * 16 + 8 * hi];
        *(v4f*)op       = lo4;
        *(v4f*)(op + 4) = hi4;
    }
}

extern "C" void kernel_launch(void* const* d_in, const int* in_sizes, int n_in,
                              void* d_out, int out_size, void* d_ws, size_t ws_size,
                              hipStream_t stream) {
    (void)in_sizes; (void)n_in; (void)out_size; (void)d_ws; (void)ws_size;
    const float* Q = (const float*)d_in[0];
    const float* K = (const float*)d_in[1];
    const float* V = (const float*)d_in[2];
    float* O = (float*)d_out;

    dim3 grid(NS / QTILE, NQH, NB);   // (16, 32, 2)
    dim3 block(NTHREADS);             // 8 wave32
    gqa_flash_wmma<<<grid, block, 0, stream>>>(Q, K, V, O);
}
